// VolumetricNCELoss_43310450213294
// MI455X (gfx1250) — matrix operand order repaired and checked by reading, not compiled, not run on hardware
//
#include <hip/hip_runtime.h>
#include <hip/hip_bf16.h>

typedef __attribute__((ext_vector_type(16))) _Float16 v16h;
typedef __attribute__((ext_vector_type(8)))  _Float16 v8h;
typedef __attribute__((ext_vector_type(8)))  float    v8f;

#define NLAYERS 5
#define NB 4
#define NP 64        // patches
#define INV_T (1.0f / 0.07f)

struct Ptrs {
    const float* q[NLAYERS];
    const float* k[NLAYERS];
    const int*   ids[NLAYERS];
};

// per-layer constants
__device__ __forceinline__ int layerC(int l)     { const int c[5] = {32, 64, 128, 256, 256}; return c[l]; }
__device__ __forceinline__ int layerCshift(int l){ const int s[5] = {5, 6, 7, 8, 8};         return s[l]; }
__device__ __forceinline__ int layerNloc(int l)  { const int n[5] = {131072, 16384, 2048, 256, 256}; return n[l]; }
// prefix offsets (in halves) of each layer's 4*64*C staging block
__device__ __forceinline__ int layerOff(int l)   { const int o[5] = {0, 8192, 24576, 57344, 122880}; return o[l]; }

#define TOTAL_HALVES 188416           // 122880 + 256*256
#define WS_PARTIAL_OFF (TOTAL_HALVES * 2 * 2)  // bytes: two f16 matrices

// ---------------------------------------------------------------------------
// Kernel 1: gather sampled patches, L2-normalize over C, store f16 patch-major
// One block per (layer, batch). 256 threads.
// ---------------------------------------------------------------------------
__global__ __launch_bounds__(256) void gather_norm_kernel(Ptrs ptrs, _Float16* qn, _Float16* kn) {
    const int blk   = blockIdx.x;        // 0..19
    const int layer = blk >> 2;
    const int b     = blk & 3;
    const int C     = layerC(layer);
    const int csh   = layerCshift(layer);
    const int nloc  = layerNloc(layer);

    const float* __restrict__ qb = ptrs.q[layer] + (long)b * C * nloc;
    const float* __restrict__ kb = ptrs.k[layer] + (long)b * C * nloc;
    const int*   __restrict__ ids = ptrs.ids[layer];

    _Float16* qrow = qn + layerOff(layer) + b * NP * C;
    _Float16* krow = kn + layerOff(layer) + b * NP * C;

    __shared__ float redq[256], redk[256];
    __shared__ float sclq[NP], sclk[NP];
    __shared__ int   ids_s[NP];

    const int tid = threadIdx.x;
    if (tid < NP) ids_s[tid] = ids[tid];
    __syncthreads();

    // 4 threads per patch: channel-strided sum of squares
    const int p = tid >> 2, s = tid & 3;
    const int loc = ids_s[p];
    float aq = 0.f, ak = 0.f;
    for (int c = s; c < C; c += 4) {
        float x = qb[(long)c * nloc + loc]; aq += x * x;
        float y = kb[(long)c * nloc + loc]; ak += y * y;
    }
    redq[tid] = aq; redk[tid] = ak;
    __syncthreads();
    if (s == 0) {
        float nq = sqrtf(redq[tid] + redq[tid + 1] + redq[tid + 2] + redq[tid + 3]);
        float nk = sqrtf(redk[tid] + redk[tid + 1] + redk[tid + 2] + redk[tid + 3]);
        sclq[p] = 1.0f / fmaxf(nq, 1e-12f);   // F.normalize: x / max(||x||, eps)
        sclk[p] = 1.0f / fmaxf(nk, 1e-12f);
    }
    __syncthreads();

    // write normalized f16, patch-major [p][C] (rows contiguous -> WMMA frags
    // become two 16B loads per lane)
    const int total = NP << csh;
    for (int idx = tid; idx < total; idx += 256) {
        const int pp = idx >> csh;
        const int cc = idx & (C - 1);
        const int l2 = ids_s[pp];
        qrow[idx] = (_Float16)(qb[(long)cc * nloc + l2] * sclq[pp]);
        krow[idx] = (_Float16)(kb[(long)cc * nloc + l2] * sclk[pp]);
    }
}

// ---------------------------------------------------------------------------
// Kernel 2: 64x64xC Gram via V_WMMA_F32_16X16X32_F16 + InfoNCE row losses.
// One block per (layer, batch). 256 threads = 8 wave32s, 2 tiles per wave.
// ---------------------------------------------------------------------------
__global__ __launch_bounds__(256) void gemm_loss_kernel(const _Float16* __restrict__ qn,
                                                        const _Float16* __restrict__ kn,
                                                        float* __restrict__ partial) {
    const int blk   = blockIdx.x;
    const int layer = blk >> 2;
    const int b     = blk & 3;
    const int C     = layerC(layer);

    const _Float16* __restrict__ qb = qn + layerOff(layer) + b * NP * C;
    const _Float16* __restrict__ kb = kn + layerOff(layer) + b * NP * C;

    __shared__ float G[64][65];     // +1 pad to dodge bank conflicts
    __shared__ float red[256];
    __shared__ float lrow[64];

    const int tid  = threadIdx.x;
    const int wave = tid >> 5;
    const int lane = tid & 31;
    const int ln   = lane & 15;     // N / M-row selector within half-wave
    const int hi   = lane >> 4;     // which half of the wave

    // Each wave computes tiles t = 2*wave, 2*wave+1 of the 4x4 tile grid.
    for (int t = wave * 2; t < wave * 2 + 2; ++t) {
        const int tm = t >> 2, tn = t & 3;
        v8f acc = {};
        const _Float16* arow = qb + (tm * 16 + ln) * C;   // A: M = tm*16+ln
        const _Float16* brow = kb + (tn * 16 + ln) * C;   // B: N = tn*16+ln
        for (int kc = 0; kc < C; kc += 32) {
            // A 16x32 f16 layout: lane<16 holds K {0..7, 16..23}, lane>=16 K {8..15, 24..31}
            v8h a_lo = *(const v8h*)(arow + kc + hi * 8);
            v8h a_hi = *(const v8h*)(arow + kc + 16 + hi * 8);
            // B 32x16 f16 layout: lane<16 holds K {0..7, 16..23}, lane>=16 K {8..15, 24..31}
            v8h b_lo = *(const v8h*)(brow + kc + hi * 8);
            v8h b_hi = *(const v8h*)(brow + kc + 16 + hi * 8);
            v16h A, Bm;
            for (int i = 0; i < 8; ++i) {
                A[i] = a_lo[i];  A[i + 8] = a_hi[i];
                Bm[i] = b_lo[i]; Bm[i + 8] = b_hi[i];
            }
            acc = __builtin_amdgcn_wmma_f32_16x16x32_f16(
                /*neg_a=*/false, A, /*neg_b=*/false, Bm,
                /*c_mod=*/(short)0, acc, /*reuse_a=*/false, /*reuse_b=*/false);
        }
        // D layout: VGPR v, lane<16 -> M = v, lane>=16 -> M = v+8; N = lane%16
        for (int v = 0; v < 8; ++v)
            G[tm * 16 + v + hi * 8][tn * 16 + ln] = acc[v];
    }
    __syncthreads();

    // InfoNCE per row: l_pos = G[p][p]; denominator = sum_q exp(G[p][q]/T)
    // (masked diagonal of l_neg is exactly replaced by the positive slot).
    const int p = tid >> 2, s = tid & 3;
    float m = -1e30f;
    for (int qi = s; qi < 64; qi += 4) m = fmaxf(m, G[p][qi]);
    red[tid] = m;
    __syncthreads();
    m = fmaxf(fmaxf(red[p * 4], red[p * 4 + 1]), fmaxf(red[p * 4 + 2], red[p * 4 + 3]));
    __syncthreads();
    float sum = 0.f;
    for (int qi = s; qi < 64; qi += 4) sum += expf((G[p][qi] - m) * INV_T);
    red[tid] = sum;
    __syncthreads();
    if (s == 0) {
        float tot = red[tid] + red[tid + 1] + red[tid + 2] + red[tid + 3];
        lrow[p] = (m - G[p][p]) * INV_T + logf(tot);
    }
    __syncthreads();
    if (tid == 0) {
        float acc = 0.f;
        for (int i = 0; i < 64; ++i) acc += lrow[i];
        partial[blk] = acc;
    }
}

// ---------------------------------------------------------------------------
// Kernel 3: deterministic fixed-order final reduction.
// ---------------------------------------------------------------------------
__global__ void finalize_kernel(const float* __restrict__ partial, float* __restrict__ out) {
    if (threadIdx.x == 0 && blockIdx.x == 0) {
        float acc = 0.f;
        for (int i = 0; i < NLAYERS * NB; ++i) acc += partial[i];
        out[0] = acc / (float)(NP * NB * NLAYERS);   // mean over (b,p), then layers
    }
}

extern "C" void kernel_launch(void* const* d_in, const int* in_sizes, int n_in,
                              void* d_out, int out_size, void* d_ws, size_t ws_size,
                              hipStream_t stream) {
    (void)in_sizes; (void)n_in; (void)out_size; (void)ws_size;
    Ptrs ptrs;
    // setup_inputs() dict order: feat_q_i, feat_k_i, ids_i per layer
    for (int l = 0; l < NLAYERS; ++l) {
        ptrs.q[l]   = (const float*)d_in[3 * l + 0];
        ptrs.k[l]   = (const float*)d_in[3 * l + 1];
        ptrs.ids[l] = (const int*)  d_in[3 * l + 2];
    }
    _Float16* qn = (_Float16*)d_ws;
    _Float16* kn = qn + TOTAL_HALVES;
    float* partial = (float*)((char*)d_ws + WS_PARTIAL_OFF);

    gather_norm_kernel<<<NLAYERS * NB, 256, 0, stream>>>(ptrs, qn, kn);
    gemm_loss_kernel<<<NLAYERS * NB, 256, 0, stream>>>(qn, kn, partial);
    finalize_kernel<<<1, 32, 0, stream>>>(partial, (float*)d_out);
}
